// MoELayer_36507222016560
// MI455X (gfx1250) — matrix operand order repaired and checked by reading, not compiled, not run on hardware
//
#include <hip/hip_runtime.h>
#include <hip/hip_bf16.h>
#include <math.h>

// ---------------------------------------------------------------------------
// MoE FFN for MI455X (gfx1250, wave32, WMMA).
// Routed top-2 (137 GFLOP f16) instead of dense-over-experts (550 GFLOP).
// Fused GEMM1 -> GELU -> GEMM2 per 16-token tile keeps h in LDS (no 64MB
// activation round trip). Weights converted once to f16 transposed copies so
// WMMA B fragments are contiguous 32B loads. x converted once to f16 so the
// tile staging can use GLOBAL_LOAD_ASYNC_TO_LDS_B128 (ASYNCcnt path).
// Output scatter uses f32 atomic add: exactly 2 commutative contributions
// per token -> bitwise deterministic.
// ---------------------------------------------------------------------------

typedef __attribute__((ext_vector_type(16))) _Float16 v16h;
typedef __attribute__((ext_vector_type(8)))  _Float16 v8h;
typedef __attribute__((ext_vector_type(4)))  _Float16 v4h;
typedef __attribute__((ext_vector_type(8)))  float    v8f;

#define D_MODEL 1024
#define D_FF    4096
#define NEXP    8
#define T_TOK   4096
#define TM      16      // tokens per tile
#define FC      128     // D_FF chunk per fused iteration
#define XSTR    1032    // padded LDS stride (halves) for x tile
#define HSTR    136     // padded LDS stride (halves) for h chunk

// ---------------------------- zero init ------------------------------------
__global__ void k_zero(float* __restrict__ out, int n, int* __restrict__ cnt) {
  int i = blockIdx.x * blockDim.x + threadIdx.x;
  if (i < n) out[i] = 0.0f;
  if (i < NEXP) cnt[i] = 0;
}

// ---------------------- x: f32 -> f16 copy (for async staging) --------------
__global__ void k_xh(const float* __restrict__ x, _Float16* __restrict__ xh) {
  int i = blockIdx.x * blockDim.x + threadIdx.x;   // over float4s
  if (i < T_TOK * D_MODEL / 4) {
    const float4 f = ((const float4*)x)[i];
    v4h h = { (_Float16)f.x, (_Float16)f.y, (_Float16)f.z, (_Float16)f.w };
    *(v4h*)(xh + (size_t)i * 4) = h;
  }
}

// ------------------- f32 -> f16 transposed weight prep ----------------------
// W1[e][k][f] -> W1T[e][f][k];  W2[e][f][n] -> W2T[e][n][f]
__global__ void k_prep(const float* __restrict__ W1, const float* __restrict__ W2,
                       _Float16* __restrict__ W1T, _Float16* __restrict__ W2T) {
  __shared__ float tile[32][33];
  const int which = blockIdx.z & 1;
  const int e     = blockIdx.z >> 1;
  int R, C; const float* src; _Float16* dst;
  if (which == 0) { R = D_MODEL; C = D_FF;
                    src = W1 + (size_t)e * D_MODEL * D_FF;
                    dst = W1T + (size_t)e * D_MODEL * D_FF; }
  else            { R = D_FF; C = D_MODEL;
                    src = W2 + (size_t)e * D_MODEL * D_FF;
                    dst = W2T + (size_t)e * D_MODEL * D_FF; }
  const int c0 = blockIdx.x * 32, r0 = blockIdx.y * 32;
  if (c0 >= C || r0 >= R) return;
  const int tx = threadIdx.x & 31, ty0 = threadIdx.x >> 5;
  for (int ty = ty0; ty < 32; ty += 8)
    tile[ty][tx] = src[(size_t)(r0 + ty) * C + c0 + tx];
  __syncthreads();
  for (int ty = ty0; ty < 32; ty += 8)
    dst[(size_t)(c0 + ty) * R + r0 + tx] = (_Float16)tile[tx][ty];
}

// ----------------------------- gating --------------------------------------
__global__ void k_gate(const float* __restrict__ x, const float* __restrict__ Wg,
                       int* __restrict__ cnt, int* __restrict__ idxL,
                       float* __restrict__ wgtL) {
  const int wid = threadIdx.x >> 5, lane = threadIdx.x & 31;
  const int t = blockIdx.x * 8 + wid;
  if (t >= T_TOK) return;
  float acc[NEXP];
#pragma unroll
  for (int e = 0; e < NEXP; ++e) acc[e] = 0.0f;
  const float* xr = x + (size_t)t * D_MODEL;
  for (int d = lane; d < D_MODEL; d += 32) {
    const float xv = xr[d];
    const float* wr = Wg + d * NEXP;
#pragma unroll
    for (int e = 0; e < NEXP; ++e) acc[e] += xv * wr[e];
  }
#pragma unroll
  for (int e = 0; e < NEXP; ++e)
    for (int off = 16; off; off >>= 1) acc[e] += __shfl_xor(acc[e], off, 32);
  if (lane == 0) {
    float mx = acc[0];
#pragma unroll
    for (int e = 1; e < NEXP; ++e) mx = fmaxf(mx, acc[e]);
    float p[NEXP], s = 0.0f;
#pragma unroll
    for (int e = 0; e < NEXP; ++e) { p[e] = expf(acc[e] - mx); s += p[e]; }
#pragma unroll
    for (int e = 0; e < NEXP; ++e) p[e] /= s;
    int i0 = 0;
#pragma unroll
    for (int e = 1; e < NEXP; ++e) if (p[e] > p[i0]) i0 = e;
    int i1 = (i0 == 0) ? 1 : 0;
#pragma unroll
    for (int e = 0; e < NEXP; ++e) if (e != i0 && p[e] > p[i1]) i1 = e;
    const float sn = p[i0] + p[i1] + 1e-8f;
    const float v0 = p[i0] / sn, v1 = p[i1] / sn;
    int s0 = atomicAdd(&cnt[i0], 1);
    idxL[i0 * T_TOK + s0] = t; wgtL[i0 * T_TOK + s0] = v0;
    int s1 = atomicAdd(&cnt[i1], 1);
    idxL[i1 * T_TOK + s1] = t; wgtL[i1 * T_TOK + s1] = v1;
  }
}

// --------------------------- fused expert FFN -------------------------------
__global__ __launch_bounds__(256) void k_ffn(
    const _Float16* __restrict__ xh,
    const _Float16* __restrict__ W1T, const float* __restrict__ b1,
    const _Float16* __restrict__ W2T, const float* __restrict__ b2,
    const int* __restrict__ cnt, const int* __restrict__ idxL,
    const float* __restrict__ wgtL, float* __restrict__ out) {
  __shared__ alignas(16) _Float16 XS[TM * XSTR];   // x tile f16, padded
  __shared__ alignas(16) _Float16 HS[TM * HSTR];   // h chunk f16, padded
  __shared__ int   stok[TM];
  __shared__ float swv[TM];

  const int MT = T_TOK / TM;                 // 256 tiles per expert (max)
  const int e  = blockIdx.x / MT;
  const int m  = blockIdx.x % MT;
  const int ne = cnt[e];
  const int row0 = m * TM;
  if (row0 >= ne) return;                    // tile beyond this expert's count

  const int tid = threadIdx.x;
  if (tid < TM) {
    const int slot = row0 + tid;
    if (slot < ne) { stok[tid] = idxL[e * T_TOK + slot];
                     swv[tid]  = wgtL[e * T_TOK + slot]; }
    else           { stok[tid] = 0; swv[tid] = 0.0f; }
  }
  __syncthreads();

  // ---- stage x tile into LDS via CDNA5 async copies (ASYNCcnt path) ----
  // 16 rows x 128 x 16B chunks; per-lane LDS offsets keep the padded layout.
  for (int i = tid; i < TM * (D_MODEL / 8); i += 256) {
    const int r = i >> 7;            // row (0..15)
    const int c = i & 127;           // 16-byte chunk within row
    const unsigned ldsOff =
        (unsigned)(unsigned long long)(const void*)&XS[r * XSTR + c * 8];
    const _Float16* g = xh + (size_t)stok[r] * D_MODEL + c * 8;
    asm volatile("global_load_async_to_lds_b128 %0, %1, off"
                 :: "v"(ldsOff), "v"(g) : "memory");
  }
  asm volatile("s_wait_asynccnt 0" ::: "memory");
  __syncthreads();

  const int wid = tid >> 5, lane = tid & 31;
  const int hi = lane >> 4, lr = lane & 15;

  v8f zero = {};
  v8f acc[8];                                // this wave's 16x128 output cols
#pragma unroll
  for (int j = 0; j < 8; ++j) acc[j] = zero;

  const _Float16* W1e = W1T + (size_t)e * D_MODEL * D_FF;
  const _Float16* W2e = W2T + (size_t)e * D_MODEL * D_FF;

  for (int fc = 0; fc < D_FF; fc += FC) {
    // ---- phase 1: h[:, f0..f0+15] = gelu(x @ W1 + b1), per wave ----
    const int f0 = fc + wid * 16;
    const _Float16* bcol = W1e + (size_t)(f0 + lr) * D_MODEL;  // B^T row
    if (fc + FC < D_FF)
      __builtin_prefetch(W1e + (size_t)(f0 + FC + lr) * D_MODEL, 0, 1);
    v8f hc = zero;
#pragma unroll 2
    for (int k = 0; k < D_MODEL; k += 32) {
      v16h a, b;
      const v8h alo = *(const v8h*)&XS[lr * XSTR + k + hi * 16];
      const v8h ahi = *(const v8h*)&XS[lr * XSTR + k + 8 + hi * 16];
      const v16h bv = *(const v16h*)(bcol + k + hi * 16);
#pragma unroll
      for (int q = 0; q < 8; ++q) { a[q] = alo[q]; a[8 + q] = ahi[q]; }
      b = bv;
      hc = __builtin_amdgcn_wmma_f32_16x16x32_f16(false, a, false, b,
                                                  (short)0, hc, false, false);
    }
    const float bv1 = b1[e * D_FF + f0 + lr];
#pragma unroll
    for (int q = 0; q < 8; ++q) {
      const int mm = q + 8 * hi;
      float h = hc[q] + bv1;
      h = 0.5f * h * (1.0f + erff(h * 0.70710678118f));   // exact GELU
      HS[mm * HSTR + wid * 16 + lr] = (_Float16)h;
    }
    __syncthreads();

    // ---- phase 2: acc += h_chunk @ W2[fc:fc+128, wave's 128 cols] ----
#pragma unroll
    for (int kk = 0; kk < FC; kk += 32) {
      v16h a2;
      const v8h alo = *(const v8h*)&HS[lr * HSTR + kk + hi * 16];
      const v8h ahi = *(const v8h*)&HS[lr * HSTR + kk + 8 + hi * 16];
#pragma unroll
      for (int q = 0; q < 8; ++q) { a2[q] = alo[q]; a2[8 + q] = ahi[q]; }
#pragma unroll
      for (int j = 0; j < 8; ++j) {
        const int n = wid * 128 + j * 16 + lr;
        const v16h b = *(const v16h*)(W2e + (size_t)n * D_FF + fc + kk + hi * 16);
        acc[j] = __builtin_amdgcn_wmma_f32_16x16x32_f16(false, a2, false, b,
                                                        (short)0, acc[j],
                                                        false, false);
      }
    }
    __syncthreads();   // protect HS before next phase-1 overwrite
  }

  // ---- epilogue: out[t] += w * (y + b2)  (2 commutative adds per token) ----
  const int v = (ne - row0 < TM) ? (ne - row0) : TM;
#pragma unroll
  for (int j = 0; j < 8; ++j) {
    const int n = wid * 128 + j * 16 + lr;
    const float b2v = b2[e * D_MODEL + n];
#pragma unroll
    for (int q = 0; q < 8; ++q) {
      const int mm = q + 8 * hi;
      if (mm < v) {
        const float val = swv[mm] * (acc[j][q] + b2v);
        atomicAdd(out + (size_t)stok[mm] * D_MODEL + n, val);
      }
    }
  }
}

// ----------------------------- launcher -------------------------------------
extern "C" void kernel_launch(void* const* d_in, const int* in_sizes, int n_in,
                              void* d_out, int out_size, void* d_ws, size_t ws_size,
                              hipStream_t stream) {
  const float* x  = (const float*)d_in[0];
  const float* Wg = (const float*)d_in[1];
  const float* W1 = (const float*)d_in[2];
  const float* b1 = (const float*)d_in[3];
  const float* W2 = (const float*)d_in[4];
  const float* b2 = (const float*)d_in[5];
  float* out = (float*)d_out;

  char* ws = (char*)d_ws;
  const size_t WSZ = (size_t)NEXP * D_MODEL * D_FF * sizeof(_Float16); // 64MB
  _Float16* W1T = (_Float16*)(ws);
  _Float16* W2T = (_Float16*)(ws + WSZ);
  int*      cnt  = (int*)(ws + 2 * WSZ);
  int*      idxL = (int*)(ws + 2 * WSZ + 128);
  float*    wgtL = (float*)(ws + 2 * WSZ + 128 + (size_t)NEXP * T_TOK * 4);
  _Float16* xh   = (_Float16*)(ws + 2 * WSZ + 128 + (size_t)NEXP * T_TOK * 8);

  const int nOut = T_TOK * D_MODEL;
  k_zero<<<(nOut + 255) / 256, 256, 0, stream>>>(out, nOut, cnt);
  k_xh<<<(T_TOK * D_MODEL / 4 + 255) / 256, 256, 0, stream>>>(x, xh);
  k_prep<<<dim3(128, 128, 16), 256, 0, stream>>>(W1, W2, W1T, W2T);
  k_gate<<<T_TOK / 8, 256, 0, stream>>>(x, Wg, cnt, idxL, wgtL);
  k_ffn<<<NEXP * (T_TOK / TM), 256, 0, stream>>>(xh, W1T, b1, W2T, b2,
                                                 cnt, idxL, wgtL, out);
}
